// TimeGradPredictionNetwork_12360915878056
// MI455X (gfx1250) — matrix operand order repaired and checked by reading, not compiled, not run on hardware
//
#include <hip/hip_runtime.h>
#include <hip/hip_bf16.h>

// ---------------------------------------------------------------- types
typedef _Float16 f16;
typedef __attribute__((ext_vector_type(16))) _Float16 v16h;
typedef __attribute__((ext_vector_type(8)))  float    v8f;

// ---------------------------------------------------------------- sizes
#define BATCH      8
#define NSAMP      16
#define BS         128           // BATCH*NSAMP
#define SEQLEN     10
#define NLOOK      20
#define NFEATS     45
#define NCTRL      27
#define TTOT       39            // SEQLEN+NLOOK+1+8
#define KOUT       8             // TTOT - SEQLEN - NLOOK - 1
#define CELLS      512
#define G3         1536          // 3*CELLS
#define CONDLEN    1287          // 10*45 + 31*27
#define CONDPAD    1312          // padded to multiple of 32
#define DSTEPS     100
#define RESCH      8
#define RESHID     64
#define OUTROWS    19            // TTOT - NLOOK

// ---------------------------------------------------------------- RNG helpers
__device__ __forceinline__ unsigned pcg(unsigned v) {
    v = v * 747796405u + 2891336453u;
    unsigned w = ((v >> ((v >> 28u) + 4u)) ^ v) * 277803737u;
    return (w >> 22u) ^ w;
}
__device__ __forceinline__ float u01(unsigned v) {
    return (float)(v >> 8) * (1.0f / 16777216.0f) + (0.5f / 16777216.0f);
}
__device__ __forceinline__ float gaussf(unsigned s) {
    float u1 = u01(pcg(s));
    float u2 = u01(pcg(s ^ 0x9E3779B9u));
    return sqrtf(-2.0f * logf(u1)) * cosf(6.28318530717958647f * u2);
}

// ---------------------------------------------------------------- misc kernels
__global__ void k_zero_f32(float* p, int n) {
    int i = blockIdx.x * blockDim.x + threadIdx.x;
    if (i < n) p[i] = 0.0f;
}

// f32 -> f16 with optional K padding (dst is N x Kp, src is N x K)
__global__ void k_f32_to_f16_pad(const float* __restrict__ src, f16* __restrict__ dst,
                                 int N, int K, int Kp) {
    int i = blockIdx.x * blockDim.x + threadIdx.x;
    if (i >= N * Kp) return;
    int r = i / Kp, c = i - r * Kp;
    dst[i] = (f16)((c < K) ? src[r * K + c] : 0.0f);
}

// diffusion schedule tables: [0..99]=sqrt(1/acp) [100..]=sqrt(1/acp-1)
// [200..]=postC1 [300..]=postC2 [400..]=postLogVar
__global__ void k_build_tabs(float* tabs) {
    if (threadIdx.x != 0 || blockIdx.x != 0) return;
    double acp = 1.0;
    for (int i = 0; i < DSTEPS; ++i) {
        double beta = 1e-4 + (0.1 - 1e-4) * (double)i / (double)(DSTEPS - 1);
        double acp_prev = acp;
        acp *= (1.0 - beta);
        tabs[i]       = (float)sqrt(1.0 / acp);
        tabs[100 + i] = (float)sqrt(1.0 / acp - 1.0);
        tabs[200 + i] = (float)(beta * sqrt(acp_prev) / (1.0 - acp));
        tabs[300 + i] = (float)((1.0 - acp_prev) * sqrt(1.0 - beta) / (1.0 - acp));
        double pv = beta * (1.0 - acp_prev) / (1.0 - acp);
        tabs[400 + i] = (float)log(pv > 1e-20 ? pv : 1e-20);
    }
}

// time embedding MLP, all 100 steps: e_all[t][64]
__global__ void k_emb_proj(const float* __restrict__ table,  // (100,32)
                           const float* __restrict__ w1,     // (32,64)
                           const float* __restrict__ b1,
                           const float* __restrict__ w2,     // (64,64)
                           const float* __restrict__ b2,
                           float* __restrict__ e_all) {
    int t = blockIdx.x, j = threadIdx.x;           // 100 blocks x 64 threads
    __shared__ float hid[RESHID];
    float a = b1[j];
    for (int i = 0; i < 32; ++i) a += table[t * 32 + i] * w1[i * RESHID + j];
    hid[j] = a / (1.0f + expf(-a));                // silu
    __syncthreads();
    float o = b2[j];
    for (int i = 0; i < RESHID; ++i) o += hid[i] * w2[i * RESHID + j];
    e_all[t * RESHID + j] = o / (1.0f + expf(-o)); // silu
}

// assemble GRU conditioning row (f16, padded). k<0 => init mode (16 rows, autoreg=0,
// ctrl slice [0:31], rows>=8 zeroed); k>=0 => 128 rows, ctrl slice [k+1:k+32].
__global__ void k_build_cond(const float* __restrict__ ctrl,     // (8,39,27)
                             const float* __restrict__ autoreg,  // (8,10,45)
                             f16* __restrict__ out, int k) {
    int nrows = (k < 0) ? 16 : BS;
    int i = blockIdx.x * blockDim.x + threadIdx.x;
    if (i >= nrows * CONDPAD) return;
    int row = i / CONDPAD, col = i - row * CONDPAD;
    float v = 0.0f;
    if (k < 0) {
        if (row < 8 && col >= 450 && col < CONDLEN) {
            int c2 = col - 450, tt = c2 / NCTRL, cc = c2 - tt * NCTRL;
            v = ctrl[(row * TTOT + tt) * NCTRL + cc];
        }
    } else {
        int b = row >> 4;
        if (col < 450) {
            v = autoreg[b * 450 + col];
        } else if (col < CONDLEN) {
            int c2 = col - 450, tt = c2 / NCTRL, cc = c2 - tt * NCTRL;
            v = ctrl[(b * TTOT + (k + 1 + tt)) * NCTRL + cc];
        }
    }
    out[row * CONDPAD + col] = (f16)v;
}

// ---------------------------------------------------------------- WMMA GEMM
// C[m,n] = sum_k A[m,k]*B[n,k] + bias[n]   (B = weight, row-major N x ldb)
// One wave computes a 16x64 strip (4 N-tiles) so the A fragment is loaded once
// and reused by 4 back-to-back v_wmma_f32_16x16x32_f16 ops per K-tile.
__global__ void __launch_bounds__(32)
k_gemm_wmma(const f16* __restrict__ A, int lda,
            const f16* __restrict__ B, int ldb,
            const float* __restrict__ bias,
            float* __restrict__ C, int ldc, int ktiles) {
    const int lane  = threadIdx.x;
    const int halfw = lane >> 4;       // 0 or 1
    const int mn    = lane & 15;
    const int Mbase = blockIdx.y * 16;
    const int Nbase = blockIdx.x * 64;
    const f16* arow = A + (size_t)(Mbase + mn) * lda;
    const f16* brow = B + (size_t)(Nbase + mn) * ldb;
    const size_t bstep = (size_t)16 * ldb;
    v8f acc0 = {}, acc1 = {}, acc2 = {}, acc3 = {};
    for (int kt = 0; kt < ktiles; ++kt) {
        const int kb = kt * 32;
        // near-cache (WGP-scope) prefetch of the next K-tile
        __builtin_prefetch(arow + kb + 32, 0, 3);
        __builtin_prefetch(brow + kb + 32, 0, 3);
        __builtin_prefetch(brow + bstep + kb + 32, 0, 3);
        __builtin_prefetch(brow + 2 * bstep + kb + 32, 0, 3);
        __builtin_prefetch(brow + 3 * bstep + kb + 32, 0, 3);
        union { v16h v; unsigned u[8]; } a, b0, b1, b2, b3;
#pragma unroll
        for (int j = 0; j < 8; ++j) {
            // A 16x32 f16: lanes 0-15 K=0..15, lanes 16-31 K=16..31 (matches B)
            int ka = kb + 16 * halfw + 2 * j;
            a.u[j] = *(const unsigned*)(arow + ka);
            // B 32x16 f16: lanes 0-15 K=0..15, lanes 16-31 K=16..31
            int kB = kb + 16 * halfw + 2 * j;
            b0.u[j] = *(const unsigned*)(brow + kB);
            b1.u[j] = *(const unsigned*)(brow + bstep + kB);
            b2.u[j] = *(const unsigned*)(brow + 2 * bstep + kB);
            b3.u[j] = *(const unsigned*)(brow + 3 * bstep + kB);
        }
        acc0 = __builtin_amdgcn_wmma_f32_16x16x32_f16(false, a.v, false, b0.v, (short)0, acc0, false, false);
        acc1 = __builtin_amdgcn_wmma_f32_16x16x32_f16(false, a.v, false, b1.v, (short)0, acc1, false, false);
        acc2 = __builtin_amdgcn_wmma_f32_16x16x32_f16(false, a.v, false, b2.v, (short)0, acc2, false, false);
        acc3 = __builtin_amdgcn_wmma_f32_16x16x32_f16(false, a.v, false, b3.v, (short)0, acc3, false, false);
    }
    v8f accs[4] = {acc0, acc1, acc2, acc3};
#pragma unroll
    for (int tt = 0; tt < 4; ++tt) {
        const int col = Nbase + 16 * tt + mn;
        const float bv = bias ? bias[col] : 0.0f;
#pragma unroll
        for (int r = 0; r < 8; ++r) {
            int row = Mbase + r + 8 * halfw;   // C layout: VGPR r -> M=r / r+8
            C[(size_t)row * ldc + col] = accs[tt][r] + bv;
        }
    }
}

// GRU gate fusion: consumes gi/gh (bias already fused), updates h, emits f16 copy
__global__ void k_gru_combine(const float* __restrict__ gi, const float* __restrict__ gh,
                              float* __restrict__ h, f16* __restrict__ h16, int bs) {
    int i = blockIdx.x * blockDim.x + threadIdx.x;
    if (i >= bs * CELLS) return;
    int row = i / CELLS, c = i - row * CELLS;
    const float* gir = gi + (size_t)row * G3;
    const float* ghr = gh + (size_t)row * G3;
    float r = 1.0f / (1.0f + expf(-(gir[c] + ghr[c])));
    float z = 1.0f / (1.0f + expf(-(gir[CELLS + c] + ghr[CELLS + c])));
    float n = tanhf(gir[2 * CELLS + c] + r * ghr[2 * CELLS + c]);
    float hl = (1.0f - z) * n + z * h[i];
    h[i] = hl;
    if (h16) h16[i] = (f16)hl;
}

__global__ void k_repeat_state(const float* __restrict__ hinit,  // (2,16,512) rows 0..7 valid
                               float* __restrict__ state) {      // (2,128,512)
    int i = blockIdx.x * blockDim.x + threadIdx.x;
    if (i >= 2 * BS * CELLS) return;
    int l = i / (BS * CELLS);
    int rem = i - l * (BS * CELLS);
    int sb = rem / CELLS, c = rem - sb * CELLS;
    state[i] = hinit[(l * 16 + (sb >> 4)) * CELLS + c];
}

// cond projection: c45 = leaky(leaky(rnn@cu_w1+b1)@cu_w2+b2), per sample row
__global__ void k_cond_proj(const float* __restrict__ rnn,   // (128,512)
                            const float* __restrict__ w1,    // (512,22)
                            const float* __restrict__ b1,
                            const float* __restrict__ w2,    // (22,45)
                            const float* __restrict__ b2,
                            float* __restrict__ c45) {
    int b = blockIdx.x, tid = threadIdx.x;  // 128 blocks x 64 threads
    __shared__ float hid[22];
    if (tid < 22) {
        float a = b1[tid];
        const float* r = rnn + (size_t)b * CELLS;
        for (int i = 0; i < CELLS; ++i) a += r[i] * w1[i * 22 + tid];
        hid[tid] = (a > 0.0f) ? a : 0.4f * a;
    }
    __syncthreads();
    if (tid < NFEATS) {
        float a = b2[tid];
        for (int i = 0; i < 22; ++i) a += hid[i] * w2[i * NFEATS + tid];
        c45[b * NFEATS + tid] = (a > 0.0f) ? a : 0.4f * a;
    }
}

__global__ void k_init_x(float* __restrict__ x, int k) {
    int i = blockIdx.x * blockDim.x + threadIdx.x;
    if (i >= BS * NFEATS) return;
    x[i] = gaussf(0x51ED270Bu ^ (unsigned)(k * 0x9E3779B9u) ^ (unsigned)(i * 0x85EBCA6Bu));
}

// ---------------------------------------------------------------- eps-net step
struct EpsBlk {
    const float *dil_w, *dil_b, *dp_w, *dp_b, *cp_w, *cp_b, *op_w, *op_b;
};
struct EpsAll {
    const float *inp_w, *inp_b, *skip_w, *skip_b, *out_w, *out_b;
    EpsBlk blk[8];
};

#define LPAD 48
__global__ void __launch_bounds__(64)
k_diff_step(float* __restrict__ x,             // (128,45), in-place
            const float* __restrict__ c45,     // (128,45)
            const float* __restrict__ e_all,   // (100,64)
            const float* __restrict__ tabs,    // (500)
            EpsAll P, int t, int k) {
    const int row = blockIdx.x, tid = threadIdx.x;
    __shared__ float sx[LPAD], sc[LPAD], se[RESHID], sds[RESCH];
    __shared__ float sh[RESCH][LPAD], sy[2 * RESCH][LPAD], ssk[RESCH][LPAD];

    if (tid < NFEATS) { sx[tid] = x[row * NFEATS + tid]; sc[tid] = c45[row * NFEATS + tid]; }
    se[tid] = e_all[t * RESHID + tid];
    __syncthreads();

    // input conv (1->8, k=1) + leaky(0.4); zero skip
    for (int idx = tid; idx < RESCH * NFEATS; idx += 64) {
        int ch = idx / NFEATS, i = idx - ch * NFEATS;
        float a = P.inp_w[ch] * sx[i] + P.inp_b[ch];
        sh[ch][i]  = (a > 0.0f) ? a : 0.4f * a;
        ssk[ch][i] = 0.0f;
    }
    __syncthreads();

    for (int bi = 0; bi < 8; ++bi) {
        const EpsBlk& B = P.blk[bi];
        const int dil = 1 << (bi & 1);
        if (tid < RESCH) {
            float a = B.dp_b[tid];
            for (int j = 0; j < RESHID; ++j) a += se[j] * B.dp_w[j * RESCH + tid];
            sds[tid] = a;
        }
        __syncthreads();
        // dilated conv (8->16, k=3, pad=dil) + cond pointwise (1->16)
        for (int idx = tid; idx < 2 * RESCH * NFEATS; idx += 64) {
            int oc = idx / NFEATS, i = idx - oc * NFEATS;
            float a = B.dil_b[oc] + B.cp_w[oc] * sc[i] + B.cp_b[oc];
            for (int ic = 0; ic < RESCH; ++ic) {
                float hv = sds[ic];
#pragma unroll
                for (int kk = 0; kk < 3; ++kk) {
                    int ii = i + (kk - 1) * dil;
                    if (ii >= 0 && ii < NFEATS)
                        a += B.dil_w[(oc * RESCH + ic) * 3 + kk] * (sh[ic][ii] + hv);
                }
            }
            sy[oc][i] = a;
        }
        __syncthreads();
        // gated activation (in place on lower half of sy)
        for (int idx = tid; idx < RESCH * NFEATS; idx += 64) {
            int ch = idx / NFEATS, i = idx - ch * NFEATS;
            float g = 1.0f / (1.0f + expf(-sy[ch][i]));
            float f = tanhf(sy[ch + RESCH][i]);
            sy[ch][i] = g * f;
        }
        __syncthreads();
        // output pointwise (8->16), split res/skip, residual update
        for (int idx = tid; idx < RESCH * NFEATS; idx += 64) {
            int ch = idx / NFEATS, i = idx - ch * NFEATS;
            float res = B.op_b[ch], s = B.op_b[ch + RESCH];
            for (int ic = 0; ic < RESCH; ++ic) {
                res += B.op_w[ch * RESCH + ic] * sy[ic][i];
                s   += B.op_w[(ch + RESCH) * RESCH + ic] * sy[ic][i];
            }
            sh[ch][i]   = (sh[ch][i] + res) * 0.70710678118654752f;
            ssk[ch][i] += s;
        }
        __syncthreads();
    }

    // skip conv (8->8, k=3, pad=1) + relu, on skip/sqrt(8)
    const float inv_sqrt8 = 0.35355339059327373f;
    for (int idx = tid; idx < RESCH * NFEATS; idx += 64) {
        int oc = idx / NFEATS, i = idx - oc * NFEATS;
        float a = P.skip_b[oc];
        for (int ic = 0; ic < RESCH; ++ic) {
#pragma unroll
            for (int kk = 0; kk < 3; ++kk) {
                int ii = i + kk - 1;
                if (ii >= 0 && ii < NFEATS)
                    a += P.skip_w[(oc * RESCH + ic) * 3 + kk] * ssk[ic][ii] * inv_sqrt8;
            }
        }
        sy[oc][i] = (a > 0.0f) ? a : 0.0f;
    }
    __syncthreads();

    // output conv (8->1, k=3, pad=1) -> eps; fused diffusion posterior update
    if (tid < NFEATS) {
        float eps = P.out_b[0];
        for (int ic = 0; ic < RESCH; ++ic) {
#pragma unroll
            for (int kk = 0; kk < 3; ++kk) {
                int ii = tid + kk - 1;
                if (ii >= 0 && ii < NFEATS) eps += P.out_w[ic * 3 + kk] * sy[ic][ii];
            }
        }
        float sra  = tabs[t], srm1 = tabs[100 + t];
        float c1   = tabs[200 + t], c2 = tabs[300 + t], plv = tabs[400 + t];
        float x0 = sra * sx[tid] - srm1 * eps;
        x0 = fminf(1.0f, fmaxf(-1.0f, x0));
        float mean = c1 * x0 + c2 * sx[tid];
        float xn = mean;
        if (t > 0) {
            float nz = gaussf(0xC2B2AE35u ^ (unsigned)(k * 0x27220A95u) ^
                              (unsigned)(t * 0x165667B1u) ^
                              (unsigned)((row * NFEATS + tid) * 0x9E3779B9u));
            xn += expf(0.5f * plv) * nz;
        }
        x[row * NFEATS + tid] = xn;
    }
}

// median over 16 samples (quantile 0.5 linear interp), actnorm, emit + ring update
__global__ void k_quantile(const float* __restrict__ x,      // (128,45)
                           const float* __restrict__ logs,   // (45)
                           const float* __restrict__ abias,  // (45)
                           float* __restrict__ out,          // (8,19,45)
                           float* __restrict__ autoreg,      // (8,10,45)
                           int k) {
    int idx = blockIdx.x * blockDim.x + threadIdx.x;
    if (idx >= BATCH * NFEATS) return;
    int b = idx / NFEATS, i = idx - b * NFEATS;
    float sc = expf(-logs[i]), bi = abias[i];
    float v[NSAMP];
    for (int j = 0; j < NSAMP; ++j) {
        float t = x[(b * NSAMP + j) * NFEATS + i] * sc - bi;
        int p = j;
        while (p > 0 && v[p - 1] > t) { v[p] = v[p - 1]; --p; }
        v[p] = t;
    }
    float q = 0.5f * (v[7] + v[8]);
    out[(b * OUTROWS + (SEQLEN + k)) * NFEATS + i] = q;
    // shift autoreg ring (each thread owns one column -> race-free)
    float col[SEQLEN];
    for (int r = 0; r < SEQLEN; ++r) col[r] = autoreg[(b * SEQLEN + r) * NFEATS + i];
    for (int r = 0; r < SEQLEN - 1; ++r) autoreg[(b * SEQLEN + r) * NFEATS + i] = col[r + 1];
    autoreg[(b * SEQLEN + SEQLEN - 1) * NFEATS + i] = q;
}

// ---------------------------------------------------------------- host
static inline size_t alignup(size_t x) { return (x + 255) & ~(size_t)255; }

extern "C" void kernel_launch(void* const* d_in, const int* in_sizes, int n_in,
                              void* d_out, int out_size, void* d_ws, size_t ws_size,
                              hipStream_t stream) {
    (void)in_sizes; (void)n_in; (void)out_size; (void)ws_size;
    // ---- inputs (setup_inputs dict leaf order)
    const float* autoreg_all = (const float*)d_in[0];  (void)autoreg_all;
    const float* control_all = (const float*)d_in[1];
    const float* w_ih0 = (const float*)d_in[2];
    const float* w_hh0 = (const float*)d_in[3];
    const float* b_ih0 = (const float*)d_in[4];
    const float* b_hh0 = (const float*)d_in[5];
    const float* w_ih1 = (const float*)d_in[6];
    const float* w_hh1 = (const float*)d_in[7];
    const float* b_ih1 = (const float*)d_in[8];
    const float* b_hh1 = (const float*)d_in[9];
    const float* emb_table = (const float*)d_in[10];
    const float* emb_w1 = (const float*)d_in[11];
    const float* emb_b1 = (const float*)d_in[12];
    const float* emb_w2 = (const float*)d_in[13];
    const float* emb_b2 = (const float*)d_in[14];
    const float* cu_w1 = (const float*)d_in[15];
    const float* cu_b1 = (const float*)d_in[16];
    const float* cu_w2 = (const float*)d_in[17];
    const float* cu_b2 = (const float*)d_in[18];
    EpsAll P;
    P.inp_w  = (const float*)d_in[19];
    P.inp_b  = (const float*)d_in[20];
    P.skip_w = (const float*)d_in[21];
    P.skip_b = (const float*)d_in[22];
    P.out_w  = (const float*)d_in[23];
    P.out_b  = (const float*)d_in[24];
    for (int bi = 0; bi < 8; ++bi) {
        P.blk[bi].dil_w = (const float*)d_in[25 + 8 * bi + 0];
        P.blk[bi].dil_b = (const float*)d_in[25 + 8 * bi + 1];
        P.blk[bi].dp_w  = (const float*)d_in[25 + 8 * bi + 2];
        P.blk[bi].dp_b  = (const float*)d_in[25 + 8 * bi + 3];
        P.blk[bi].cp_w  = (const float*)d_in[25 + 8 * bi + 4];
        P.blk[bi].cp_b  = (const float*)d_in[25 + 8 * bi + 5];
        P.blk[bi].op_w  = (const float*)d_in[25 + 8 * bi + 6];
        P.blk[bi].op_b  = (const float*)d_in[25 + 8 * bi + 7];
    }
    const float* act_logs = (const float*)d_in[89];
    const float* act_bias = (const float*)d_in[90];
    float* out = (float*)d_out;

    // ---- workspace carve-up
    char* ws = (char*)d_ws;
    size_t off = 0;
    auto take = [&](size_t bytes) { char* p = ws + off; off = alignup(off + bytes); return p; };
    f16*   wih0_h = (f16*)take((size_t)G3 * CONDPAD * 2);
    f16*   whh0_h = (f16*)take((size_t)G3 * CELLS * 2);
    f16*   wih1_h = (f16*)take((size_t)G3 * CELLS * 2);
    f16*   whh1_h = (f16*)take((size_t)G3 * CELLS * 2);
    f16*   cond_h = (f16*)take((size_t)BS * CONDPAD * 2);
    f16*   h16a   = (f16*)take((size_t)BS * CELLS * 2);
    f16*   h16b   = (f16*)take((size_t)BS * CELLS * 2);
    float* gi     = (float*)take((size_t)BS * G3 * 4);
    float* gh     = (float*)take((size_t)BS * G3 * 4);
    float* state  = (float*)take((size_t)2 * BS * CELLS * 4);   // [layer][row][cell]
    float* hinit  = (float*)take((size_t)2 * 16 * CELLS * 4);
    float* xbuf   = (float*)take((size_t)BS * NFEATS * 4);
    float* c45    = (float*)take((size_t)BS * NFEATS * 4);
    float* e_all  = (float*)take((size_t)DSTEPS * RESHID * 4);
    float* tabs   = (float*)take((size_t)500 * 4);
    float* aring  = (float*)take((size_t)BATCH * SEQLEN * NFEATS * 4);

    const dim3 B256(256);
    auto blocks = [](int n) { return dim3((n + 255) / 256); };

    // ---- one-time-per-call prep (deterministic)
    k_zero_f32<<<blocks(BATCH * OUTROWS * NFEATS), B256, 0, stream>>>(out, BATCH * OUTROWS * NFEATS);
    k_zero_f32<<<blocks(2 * 16 * CELLS), B256, 0, stream>>>(hinit, 2 * 16 * CELLS);
    k_zero_f32<<<blocks(BATCH * SEQLEN * NFEATS), B256, 0, stream>>>(aring, BATCH * SEQLEN * NFEATS);
    k_f32_to_f16_pad<<<blocks(G3 * CONDPAD), B256, 0, stream>>>(w_ih0, wih0_h, G3, CONDLEN, CONDPAD);
    k_f32_to_f16_pad<<<blocks(G3 * CELLS), B256, 0, stream>>>(w_hh0, whh0_h, G3, CELLS, CELLS);
    k_f32_to_f16_pad<<<blocks(G3 * CELLS), B256, 0, stream>>>(w_ih1, wih1_h, G3, CELLS, CELLS);
    k_f32_to_f16_pad<<<blocks(G3 * CELLS), B256, 0, stream>>>(w_hh1, whh1_h, G3, CELLS, CELLS);
    k_build_tabs<<<1, 1, 0, stream>>>(tabs);
    k_emb_proj<<<DSTEPS, RESHID, 0, stream>>>(emb_table, emb_w1, emb_b1, emb_w2, emb_b2, e_all);

    const int NT4 = G3 / 64;         // 24 N-strips of 64 columns
    const int KT_COND = CONDPAD / 32;
    const int KT_CELL = CELLS / 32;

    auto gru_layer = [&](const f16* A, int lda, const f16* Wih, int kt_ih,
                         const f16* Hprev16, const f16* Whh, const float* bih,
                         const float* bhh, float* hst, f16* hout16, int mtiles, int bs) {
        k_gemm_wmma<<<dim3(NT4, mtiles), 32, 0, stream>>>(A, lda, Wih, (kt_ih == KT_COND) ? CONDPAD : CELLS,
                                                          bih, gi, G3, kt_ih);
        k_gemm_wmma<<<dim3(NT4, mtiles), 32, 0, stream>>>(Hprev16, CELLS, Whh, CELLS,
                                                          bhh, gh, G3, KT_CELL);
        k_gru_combine<<<blocks(bs * CELLS), B256, 0, stream>>>(gi, gh, hst, hout16, bs);
    };

    // ---- initial GRU step (batch=8 padded to one 16-row M-tile, h=0)
    k_build_cond<<<blocks(BS * CONDPAD), B256, 0, stream>>>(control_all, aring, cond_h, -1);
    k_f32_to_f16_pad<<<blocks(16 * CELLS), B256, 0, stream>>>(hinit, h16b, 16, CELLS, CELLS);
    gru_layer(cond_h, CONDPAD, wih0_h, KT_COND, h16b, whh0_h, b_ih0, b_hh0, hinit, h16a, 1, 16);
    k_f32_to_f16_pad<<<blocks(16 * CELLS), B256, 0, stream>>>(hinit + 16 * CELLS, h16b, 16, CELLS, CELLS);
    gru_layer(h16a, CELLS, wih1_h, KT_CELL, h16b, whh1_h, b_ih1, b_hh1, hinit + 16 * CELLS, nullptr, 1, 16);
    k_repeat_state<<<blocks(2 * BS * CELLS), B256, 0, stream>>>(hinit, state);

    // ---- outer autoregressive loop
    for (int k = 0; k < KOUT; ++k) {
        k_build_cond<<<blocks(BS * CONDPAD), B256, 0, stream>>>(control_all, aring, cond_h, k);
        k_f32_to_f16_pad<<<blocks(BS * CELLS), B256, 0, stream>>>(state, h16b, BS, CELLS, CELLS);
        gru_layer(cond_h, CONDPAD, wih0_h, KT_COND, h16b, whh0_h, b_ih0, b_hh0, state, h16a, 8, BS);
        k_f32_to_f16_pad<<<blocks(BS * CELLS), B256, 0, stream>>>(state + BS * CELLS, h16b, BS, CELLS, CELLS);
        gru_layer(h16a, CELLS, wih1_h, KT_CELL, h16b, whh1_h, b_ih1, b_hh1, state + BS * CELLS, nullptr, 8, BS);

        k_cond_proj<<<BS, 64, 0, stream>>>(state + BS * CELLS, cu_w1, cu_b1, cu_w2, cu_b2, c45);
        k_init_x<<<blocks(BS * NFEATS), B256, 0, stream>>>(xbuf, k);
        for (int t = DSTEPS - 1; t >= 0; --t)
            k_diff_step<<<BS, 64, 0, stream>>>(xbuf, c45, e_all, tabs, P, t, k);
        k_quantile<<<blocks(BATCH * NFEATS), B256, 0, stream>>>(xbuf, act_logs, act_bias, out, aring, k);
    }
}
